// MOELoraInjectedLinear_68556267979240
// MI455X (gfx1250) — matrix operand order repaired and checked by reading, not compiled, not run on hardware
//
#include <hip/hip_runtime.h>
#include <hip/hip_bf16.h>

typedef __bf16 bf16_t;
typedef __attribute__((ext_vector_type(16))) __bf16 v16bf;
typedef __attribute__((ext_vector_type(8)))  __bf16 v8bf;
typedef __attribute__((ext_vector_type(8)))  float   v8f;
typedef __attribute__((ext_vector_type(4)))  float   v4f;
typedef __attribute__((__vector_size__(4 * sizeof(int)))) int v4i;

#define TOKENS 16384   // B*C = 4*4096
#define D_IN   1024
#define O_OUT  1024
#define PROJ_  256
#define KX     1056    // 1024 + 16 (packed LoRA-up) + 16 zero pad (multiple of 32)
#define G_ROWS 272     // 256 proj rows + 15 packed down rows + 1 zero row
#define P_LD   272

#define BM 128
#define BN 128
#define BK 32
#define LDT 40         // 32 + 8 pad elements -> 80B row stride, conflict-friendly

#if __has_builtin(__builtin_amdgcn_global_load_async_to_lds_b128)
#define USE_ASYNC_LDS 1
#else
#define USE_ASYNC_LDS 0
#endif

typedef __attribute__((address_space(1))) v4i* gv4i_p;   // global int4*
typedef __attribute__((address_space(3))) v4i* lv4i_p;   // LDS int4*

__device__ __forceinline__ void wait_async0() {
#if __has_builtin(__builtin_amdgcn_s_wait_asynccnt)
  __builtin_amdgcn_s_wait_asynccnt(0);
#else
  asm volatile("s_wait_asynccnt 0x0" ::: "memory");
#endif
}

// ---------------------------------------------------------------------------
// Convert x (f32) -> xb bf16, columns 0..1023 of the K=1056 padded row.
__global__ __launch_bounds__(256)
void k_convert_x(const float* __restrict__ x, bf16_t* __restrict__ xb) {
  int ci = blockIdx.x * blockDim.x + threadIdx.x;   // chunk id (8 elems)
  int n  = ci >> 7;                                 // 128 chunks per token row
  int c8 = ci & 127;
  const float* src = x + (size_t)n * D_IN + c8 * 8;
  v4f a = *(const v4f*)src;
  v4f b = *(const v4f*)(src + 4);
  v8bf o;
  o[0] = (bf16_t)a[0]; o[1] = (bf16_t)a[1]; o[2] = (bf16_t)a[2]; o[3] = (bf16_t)a[3];
  o[4] = (bf16_t)b[0]; o[5] = (bf16_t)b[1]; o[6] = (bf16_t)b[2]; o[7] = (bf16_t)b[3];
  *(v8bf*)(xb + (size_t)n * KX + c8 * 8) = o;
}

// ---------------------------------------------------------------------------
// Pack Wb[o][0..1023] = W, Wb[o][1024..1038] = packed Wu (r=1,2,4,8), rest 0.
__global__ __launch_bounds__(256)
void k_pack_W(const float* __restrict__ W,
              const float* __restrict__ Wu0, const float* __restrict__ Wu1,
              const float* __restrict__ Wu2, const float* __restrict__ Wu3,
              bf16_t* __restrict__ Wb) {
  int idx = blockIdx.x * blockDim.x + threadIdx.x;
  if (idx >= O_OUT * KX) return;
  int o = idx / KX;
  int c = idx - o * KX;
  float v = 0.0f;
  if (c < D_IN) {
    v = W[(size_t)o * D_IN + c];
  } else {
    int j = c - D_IN;
    if (j == 0)       v = Wu0[o];
    else if (j < 3)   v = Wu1[o * 2 + (j - 1)];
    else if (j < 7)   v = Wu2[o * 4 + (j - 3)];
    else if (j < 15)  v = Wu3[o * 8 + (j - 7)];
  }
  Wb[idx] = (bf16_t)v;
}

// ---------------------------------------------------------------------------
// Pack G[r][k]: rows 0..255 = Wp, rows 256..270 = packed Wd, row 271 = 0.
__global__ __launch_bounds__(256)
void k_pack_G(const float* __restrict__ Wp,
              const float* __restrict__ Wd0, const float* __restrict__ Wd1,
              const float* __restrict__ Wd2, const float* __restrict__ Wd3,
              bf16_t* __restrict__ G) {
  int idx = blockIdx.x * blockDim.x + threadIdx.x;
  if (idx >= G_ROWS * D_IN) return;
  int r = idx / D_IN;
  int c = idx - r * D_IN;
  float v = 0.0f;
  if (r < PROJ_) {
    v = Wp[(size_t)r * D_IN + c];
  } else {
    int q = r - PROJ_;
    if (q == 0)      v = Wd0[c];
    else if (q < 3)  v = Wd1[(q - 1) * D_IN + c];
    else if (q < 7)  v = Wd2[(q - 3) * D_IN + c];
    else if (q < 15) v = Wd3[(q - 7) * D_IN + c];
  }
  G[idx] = (bf16_t)v;
}

// ---------------------------------------------------------------------------
__device__ __forceinline__ v16bf mkfrag(const bf16_t* p0, const bf16_t* p1) {
  v8bf lo = *(const v8bf*)p0;
  v8bf hi = *(const v8bf*)p1;
  return __builtin_shufflevector(lo, hi,
      0, 1, 2, 3, 4, 5, 6, 7, 8, 9, 10, 11, 12, 13, 14, 15);
}

// Tiled bf16 WMMA GEMM:  Dm[M,Nn] = A[M,K] * B[Nn,K]^T  (both K-major, f32 out)
// 256 threads = 8 waves; block tile 128x128, K step 32.
// Wave tile 32x64: 2 A-frags x 4 B-frags -> 8 WMMAs, better frag reuse.
// Double-buffered LDS fed by gfx1250 async-to-LDS loads (ASYNCcnt) when
// available; falls back to staged global->VGPR->ds_store otherwise.
__global__ __launch_bounds__(256)
void k_gemm_bf16(const bf16_t* __restrict__ A, const bf16_t* __restrict__ B,
                 float* __restrict__ Dm,
                 int M, int Nn, int K, int lda, int ldb, int ldd) {
#if USE_ASYNC_LDS
  __shared__ bf16_t sA[2][BM * LDT];
  __shared__ bf16_t sB[2][BN * LDT];
#else
  __shared__ bf16_t sA[1][BM * LDT];
  __shared__ bf16_t sB[1][BN * LDT];
#endif

  const int tid  = threadIdx.x;
  const int w    = tid >> 5;
  const int lane = tid & 31;
  const int half = lane >> 4;
  const int ml   = lane & 15;
  const int tileM = blockIdx.y * BM;
  const int tileN = blockIdx.x * BN;
  const int wm = (w & 3) * 32;    // wave row base within block tile
  const int wn = (w >> 2) * 64;   // wave col base within block tile

  v8f acc[2][4] = {};

#if USE_ASYNC_LDS
  // ---- async double-buffered pipeline -------------------------------------
  auto load_tile = [&](int k0, int bsel) {
#pragma unroll
    for (int it = 0; it < 2; ++it) {
      int idx = it * 2048 + tid * 8;
      int r = idx >> 5;
      int c = idx & 31;
      const bf16_t* ga = A + (size_t)(tileM + r) * lda + k0 + c;
      __builtin_amdgcn_global_load_async_to_lds_b128(
          (gv4i_p)ga, (lv4i_p)(&sA[bsel][r * LDT + c]), 0, 0);
      int br = tileN + r;
      if (br >= Nn) br = Nn - 1;   // clamp: OOB columns are never stored
      const bf16_t* gb = B + (size_t)br * ldb + k0 + c;
      __builtin_amdgcn_global_load_async_to_lds_b128(
          (gv4i_p)gb, (lv4i_p)(&sB[bsel][r * LDT + c]), 0, 0);
    }
  };

  const int nt = K / BK;
  load_tile(0, 0);
  for (int t = 0; t < nt; ++t) {
    const int bsel = t & 1;
    wait_async0();          // own tile-t async writes to LDS complete
    __syncthreads();        // everyone's tile-t in LDS; prev reads finished
    if (t + 1 < nt) load_tile((t + 1) * BK, 1 - bsel);   // overlap with WMMA

    const bf16_t* baseA = sA[bsel];
    const bf16_t* baseB = sB[bsel];
#else
  // ---- single-buffer fallback --------------------------------------------
  for (int k0 = 0; k0 < K; k0 += BK) {
    __syncthreads();
#pragma unroll
    for (int it = 0; it < 2; ++it) {
      int idx = it * 2048 + tid * 8;
      int r = idx >> 5;
      int c = idx & 31;
      v8bf av = *(const v8bf*)(A + (size_t)(tileM + r) * lda + k0 + c);
      *(v8bf*)(&sA[0][r * LDT + c]) = av;
      int br = tileN + r;
      v8bf bv = {};
      if (br < Nn) bv = *(const v8bf*)(B + (size_t)br * ldb + k0 + c);
      *(v8bf*)(&sB[0][r * LDT + c]) = bv;
    }
    __syncthreads();
    const bf16_t* baseA = sA[0];
    const bf16_t* baseB = sB[0];
#endif

    // Load all fragments first so DS waits can be grouped, then chain WMMAs.
    v16bf af[2], bfr[4];
#pragma unroll
    for (int mi = 0; mi < 2; ++mi) {
      const bf16_t* p = &baseA[(wm + mi * 16 + ml) * LDT + half * 8];
      af[mi] = mkfrag(p, p + 16);      // K = half*8+e  /  16+half*8+(e-8)
    }
#pragma unroll
    for (int ci = 0; ci < 4; ++ci) {
      const bf16_t* p = &baseB[(wn + ci * 16 + ml) * LDT + half * 8];
      bfr[ci] = mkfrag(p, p + 16);     // K = half*8+e  /  16+half*8+(e-8)
    }
#pragma unroll
    for (int mi = 0; mi < 2; ++mi)
#pragma unroll
      for (int ci = 0; ci < 4; ++ci)
        acc[mi][ci] = __builtin_amdgcn_wmma_f32_16x16x32_bf16(
            false, af[mi], false, bfr[ci], (short)0, acc[mi][ci], false, false);
  }

  // D layout: VGPR j -> row = j + half*8, col = lane%16
#pragma unroll
  for (int mi = 0; mi < 2; ++mi) {
    const int r0 = tileM + wm + mi * 16 + half * 8;
#pragma unroll
    for (int ci = 0; ci < 4; ++ci) {
      int col = tileN + wn + ci * 16 + ml;
      if (col < Nn) {
#pragma unroll
        for (int j = 0; j < 8; ++j)
          Dm[(size_t)(r0 + j) * ldd + col] = acc[mi][ci][j];
      }
    }
  }
}

// ---------------------------------------------------------------------------
// Gate: one wave per token. l2-norm proj, cosine logits vs normalized sim,
// softmax, argmax (first-max), write u = g * masked(t) as bf16 into xb cols
// 1024..1055 (zeros outside chosen expert's rank range and in pad).
__global__ __launch_bounds__(256)
void k_gate(const float* __restrict__ P, const float* __restrict__ bp,
            const float* __restrict__ sim, const float* __restrict__ temp,
            bf16_t* __restrict__ xb) {
  const int w    = threadIdx.x >> 5;
  const int lane = threadIdx.x & 31;
  const int n    = blockIdx.x * 8 + w;
  const float* prow = P + (size_t)n * P_LD;

  float pn2 = 0.f;
  float d0 = 0.f, d1 = 0.f, d2 = 0.f, d3 = 0.f;
  float s0 = 0.f, s1 = 0.f, s2 = 0.f, s3 = 0.f;
  for (int p = lane; p < PROJ_; p += 32) {
    float pv = prow[p] + bp[p];
    pn2 += pv * pv;
    float m0 = sim[p * 4 + 0], m1 = sim[p * 4 + 1];
    float m2 = sim[p * 4 + 2], m3 = sim[p * 4 + 3];
    d0 += pv * m0; d1 += pv * m1; d2 += pv * m2; d3 += pv * m3;
    s0 += m0 * m0; s1 += m1 * m1; s2 += m2 * m2; s3 += m3 * m3;
  }
#pragma unroll
  for (int off = 16; off > 0; off >>= 1) {
    pn2 += __shfl_xor(pn2, off, 32);
    d0 += __shfl_xor(d0, off, 32); d1 += __shfl_xor(d1, off, 32);
    d2 += __shfl_xor(d2, off, 32); d3 += __shfl_xor(d3, off, 32);
    s0 += __shfl_xor(s0, off, 32); s1 += __shfl_xor(s1, off, 32);
    s2 += __shfl_xor(s2, off, 32); s3 += __shfl_xor(s3, off, 32);
  }

  const float CLAMP_MAX = 4.605170185988091f;      // log(100)
  float scale = __expf(fminf(temp[0], CLAMP_MAX));
  float rpn = 1.f / fmaxf(sqrtf(pn2), 1e-12f);
  float l0 = d0 * rpn * (1.f / fmaxf(sqrtf(s0), 1e-12f)) * scale;
  float l1 = d1 * rpn * (1.f / fmaxf(sqrtf(s1), 1e-12f)) * scale;
  float l2 = d2 * rpn * (1.f / fmaxf(sqrtf(s2), 1e-12f)) * scale;
  float l3 = d3 * rpn * (1.f / fmaxf(sqrtf(s3), 1e-12f)) * scale;

  float mx = fmaxf(fmaxf(l0, l1), fmaxf(l2, l3));
  float e0 = __expf(l0 - mx), e1 = __expf(l1 - mx);
  float e2 = __expf(l2 - mx), e3 = __expf(l3 - mx);
  float se = e0 + e1 + e2 + e3;

  int top = 0; float best = l0;
  if (l1 > best) { best = l1; top = 1; }
  if (l2 > best) { best = l2; top = 2; }
  if (l3 > best) { best = l3; top = 3; }
  float g = (top == 0 ? e0 : top == 1 ? e1 : top == 2 ? e2 : e3) / se;

  // lane j writes packed column 1024+j
  float val = 0.f;
  if (lane < 15) {
    const int offs[4] = {0, 1, 3, 7};
    const int rr[4]   = {1, 2, 4, 8};
    if (lane >= offs[top] && lane < offs[top] + rr[top]) {
      val = g * prow[PROJ_ + lane];
    }
  }
  xb[(size_t)n * KX + D_IN + lane] = (bf16_t)val;
}

// ---------------------------------------------------------------------------
extern "C" void kernel_launch(void* const* d_in, const int* in_sizes, int n_in,
                              void* d_out, int out_size, void* d_ws, size_t ws_size,
                              hipStream_t stream) {
  const float* x    = (const float*)d_in[0];
  const float* W    = (const float*)d_in[1];
  const float* Wp   = (const float*)d_in[2];
  const float* bp   = (const float*)d_in[3];
  const float* sim  = (const float*)d_in[4];
  const float* temp = (const float*)d_in[5];
  const float* Wd0  = (const float*)d_in[6];
  const float* Wu0  = (const float*)d_in[7];
  const float* Wd1  = (const float*)d_in[8];
  const float* Wu1  = (const float*)d_in[9];
  const float* Wd2  = (const float*)d_in[10];
  const float* Wu2  = (const float*)d_in[11];
  const float* Wd3  = (const float*)d_in[12];
  const float* Wu3  = (const float*)d_in[13];

  char* ws = (char*)d_ws;
  size_t off = 0;
  bf16_t* xb = (bf16_t*)(ws + off); off += (size_t)TOKENS * KX * sizeof(bf16_t);
  bf16_t* Wb = (bf16_t*)(ws + off); off += (size_t)O_OUT * KX * sizeof(bf16_t);
  bf16_t* G  = (bf16_t*)(ws + off); off += (size_t)G_ROWS * D_IN * sizeof(bf16_t);
  float*  P  = (float*)(ws + off);  off += (size_t)TOKENS * P_LD * sizeof(float);
  float* out = (float*)d_out;

  // 1) convert x to bf16 (xb cols 0..1023)
  k_convert_x<<<(TOKENS * 128) / 256, 256, 0, stream>>>(x, xb);
  // 2) pack weights
  k_pack_W<<<(O_OUT * KX + 255) / 256, 256, 0, stream>>>(W, Wu0, Wu1, Wu2, Wu3, Wb);
  k_pack_G<<<(G_ROWS * D_IN + 255) / 256, 256, 0, stream>>>(Wp, Wd0, Wd1, Wd2, Wd3, G);
  // 3) gating+down GEMM: P[N,272] = xb[:, :1024] @ G^T
  {
    dim3 grid((G_ROWS + BN - 1) / BN, TOKENS / BM);
    k_gemm_bf16<<<grid, 256, 0, stream>>>(xb, G, P, TOKENS, G_ROWS, D_IN,
                                          KX, D_IN, P_LD);
  }
  // 4) gate epilogue: writes u into xb cols 1024..1055
  k_gate<<<TOKENS / 8, 256, 0, stream>>>(P, bp, sim, temp, xb);
  // 5) fused base + LoRA GEMM: out[N,1024] = xb @ Wb^T   (K = 1056)
  {
    dim3 grid(O_OUT / BN, TOKENS / BM);
    k_gemm_bf16<<<grid, 256, 0, stream>>>(xb, Wb, out, TOKENS, O_OUT, KX,
                                          KX, KX, O_OUT);
  }
}
